// PretextLoss_9208409882650
// MI455X (gfx1250) — matrix operand order repaired and checked by reading, not compiled, not run on hardware
//
#include <hip/hip_runtime.h>

#define BS   8192
#define DIM  1024

typedef __bf16 bf16;
typedef __bf16 v16bf  __attribute__((ext_vector_type(16)));
typedef __bf16 bf16x4 __attribute__((ext_vector_type(4)));
typedef float  v8f    __attribute__((ext_vector_type(8)));
typedef float  f32x4  __attribute__((ext_vector_type(4)));
typedef unsigned int u32x4 __attribute__((ext_vector_type(4)));

union Frag { u32x4 u[2]; v16bf v; };
static_assert(sizeof(Frag) == 32, "frag size");

__device__ __forceinline__ float waveAllSum(float x) {
#pragma unroll
  for (int o = 16; o > 0; o >>= 1) x += __shfl_xor(x, o, 32);
  return x;
}

// ---------------------------------------------------------------------------
// Kernel 1: per-row L2 normalize (fp32 math), emit bf16 normalized anchor /
// negative for the WMMA GEMM, plus fp32 a_sq, n_sq, pos_dist (already /T),
// and initialize hard[] to +inf bit pattern.
// One 256-thread block (8 waves) per row; each thread owns 4 elements.
// ---------------------------------------------------------------------------
__global__ void __launch_bounds__(256)
normalize_kernel(const float* __restrict__ F,
                 bf16* __restrict__ A, bf16* __restrict__ Nb,
                 float* __restrict__ a_sq, float* __restrict__ n_sq,
                 float* __restrict__ posd, unsigned int* __restrict__ hard)
{
  __shared__ float sm[24];
  const int row  = blockIdx.x;
  const int t    = threadIdx.x;
  const int lane = t & 31;
  const int wv   = t >> 5;

  const f32x4* a4 = (const f32x4*)(F + (size_t)row * DIM);
  const f32x4* p4 = (const f32x4*)(F + ((size_t)BS + row) * DIM);
  const f32x4* n4 = (const f32x4*)(F + ((size_t)2 * BS + row) * DIM);
  f32x4 av = a4[t], pv = p4[t], nv = n4[t];

  float sa = av[0]*av[0] + av[1]*av[1] + av[2]*av[2] + av[3]*av[3];
  float sp = pv[0]*pv[0] + pv[1]*pv[1] + pv[2]*pv[2] + pv[3]*pv[3];
  float sn = nv[0]*nv[0] + nv[1]*nv[1] + nv[2]*nv[2] + nv[3]*nv[3];
  sa = waveAllSum(sa); sp = waveAllSum(sp); sn = waveAllSum(sn);
  if (lane == 0) { sm[wv] = sa; sm[8 + wv] = sp; sm[16 + wv] = sn; }
  __syncthreads();
  float Sa = 0.f, Sp = 0.f, Sn = 0.f;
#pragma unroll
  for (int w = 0; w < 8; ++w) { Sa += sm[w]; Sp += sm[8 + w]; Sn += sm[16 + w]; }

  const float ia = 1.0f / fmaxf(sqrtf(Sa), 1e-12f);
  const float ip = 1.0f / fmaxf(sqrtf(Sp), 1e-12f);
  const float in = 1.0f / fmaxf(sqrtf(Sn), 1e-12f);
  f32x4 an = av * ia, pn = pv * ip, nn = nv * in;

  bf16x4 ab, nb;
#pragma unroll
  for (int c = 0; c < 4; ++c) { ab[c] = (bf16)an[c]; nb[c] = (bf16)nn[c]; }
  *(bf16x4*)(A  + (size_t)row * DIM + 4 * t) = ab;
  *(bf16x4*)(Nb + (size_t)row * DIM + 4 * t) = nb;

  f32x4 dv = an - pn;
  float qa = an[0]*an[0] + an[1]*an[1] + an[2]*an[2] + an[3]*an[3];
  float qn = nn[0]*nn[0] + nn[1]*nn[1] + nn[2]*nn[2] + nn[3]*nn[3];
  float pd = dv[0]*dv[0] + dv[1]*dv[1] + dv[2]*dv[2] + dv[3]*dv[3];
  __syncthreads();  // sm reuse
  qa = waveAllSum(qa); qn = waveAllSum(qn); pd = waveAllSum(pd);
  if (lane == 0) { sm[wv] = qa; sm[8 + wv] = qn; sm[16 + wv] = pd; }
  __syncthreads();
  if (t == 0) {
    float Qa = 0.f, Qn = 0.f, Pd = 0.f;
#pragma unroll
    for (int w = 0; w < 8; ++w) { Qa += sm[w]; Qn += sm[8 + w]; Pd += sm[16 + w]; }
    a_sq[row] = Qa;
    n_sq[row] = Qn;
    posd[row] = Pd / 0.1f;
    hard[row] = 0x7F800000u;  // +inf
  }
}

// ---------------------------------------------------------------------------
// Kernel 2: cross = anchor_n @ negative_n^T via v_wmma_f32_16x16x32_bf16,
// fused with the per-row min of (a_sq[i] + n_sq[j] - 2*cross)/T.
// Block = 8 waves; block owns 64 anchor rows x 512 negative cols;
// each wave owns a 64x64 macro-tile (4x4 WMMA tiles), K-loop step 32.
// Fragment layouts per CDNA5 ISA 7.12.2 (wave32):
//   A 16x32 bf16:  lane holds row m=lane%16, K = 8*(lane/16)+[0..7] and +16.
//   B 32x16 bf16:  lane holds col n=lane%16, K = 8*(lane/16)+[0..7] and +16.
//   C/D 16x16 f32: lane holds N=lane%16, VGPR r -> M = r + 8*(lane/16).
// ---------------------------------------------------------------------------
__global__ void __launch_bounds__(256)
gemm_min_kernel(const bf16* __restrict__ A, const bf16* __restrict__ Nb,
                const float* __restrict__ a_sq, const float* __restrict__ n_sq,
                unsigned int* __restrict__ hard)
{
  const int lane = threadIdx.x & 31;
  const int wave = threadIdx.x >> 5;
  const int i0 = blockIdx.x * 64;
  const int j0 = blockIdx.y * 512 + wave * 64;

  const int m    = lane & 15;
  const int half = lane >> 4;

  const bf16* aB[4];
  const bf16* bB[4];
#pragma unroll
  for (int t = 0; t < 4; ++t) {
    aB[t] = A  + (size_t)(i0 + 16 * t + m) * DIM + half * 8;
    bB[t] = Nb + (size_t)(j0 + 16 * t + m) * DIM + half * 8;
  }

  v8f acc[4][4];
  const v8f vz = {0.f, 0.f, 0.f, 0.f, 0.f, 0.f, 0.f, 0.f};
#pragma unroll
  for (int ti = 0; ti < 4; ++ti)
#pragma unroll
    for (int tj = 0; tj < 4; ++tj) acc[ti][tj] = vz;

  for (int k = 0; k < DIM; k += 32) {
    Frag fa[4], fb[4];
#pragma unroll
    for (int t = 0; t < 4; ++t) {
      fa[t].u[0] = *(const u32x4*)(aB[t] + k);
      fa[t].u[1] = *(const u32x4*)(aB[t] + k + 16);
      fb[t].u[0] = *(const u32x4*)(bB[t] + k);
      fb[t].u[1] = *(const u32x4*)(bB[t] + k + 16);
    }
#pragma unroll
    for (int ti = 0; ti < 4; ++ti)
#pragma unroll
      for (int tj = 0; tj < 4; ++tj)
        acc[ti][tj] = __builtin_amdgcn_wmma_f32_16x16x32_bf16(
            false, fa[ti].v, false, fb[tj].v, (short)0, acc[ti][tj],
            false, false);
  }

  // Epilogue: e = n_sq[j] - 2*cross; min over the 64 j's of this macro-tile.
  float nsq[4];
#pragma unroll
  for (int tj = 0; tj < 4; ++tj) nsq[tj] = n_sq[j0 + 16 * tj + m];

  float mn[4][8];
#pragma unroll
  for (int ti = 0; ti < 4; ++ti)
#pragma unroll
    for (int r = 0; r < 8; ++r) {
      float v = nsq[0] - 2.0f * acc[ti][0][r];
#pragma unroll
      for (int tj = 1; tj < 4; ++tj)
        v = fminf(v, nsq[tj] - 2.0f * acc[ti][tj][r]);
      mn[ti][r] = v;
    }

  // Min across the 16 N-lanes (masks stay inside each 16-lane half).
#pragma unroll
  for (int ti = 0; ti < 4; ++ti)
#pragma unroll
    for (int r = 0; r < 8; ++r) {
      float v = mn[ti][r];
      v = fminf(v, __shfl_xor(v, 1, 32));
      v = fminf(v, __shfl_xor(v, 2, 32));
      v = fminf(v, __shfl_xor(v, 4, 32));
      v = fminf(v, __shfl_xor(v, 8, 32));
      mn[ti][r] = v;
    }

  if (m == 0) {  // lanes 0 (rows M=0..7) and 16 (rows M=8..15)
#pragma unroll
    for (int ti = 0; ti < 4; ++ti)
#pragma unroll
      for (int r = 0; r < 8; ++r) {
        const int row = i0 + 16 * ti + half * 8 + r;
        const float v = (a_sq[row] + mn[ti][r]) / 0.1f;
        atomicMin(hard + row, __float_as_uint(v));  // all values > 0
      }
  }
}

// ---------------------------------------------------------------------------
// Kernel 3: loss = mean(max(margin + pos - hard, 0)); margin = 1.0 here.
// Outputs: out[0]=loss, out[1]=mean(pos), out[2]=mean(hard).
// ---------------------------------------------------------------------------
__global__ void __launch_bounds__(256)
finalize_kernel(const float* __restrict__ posd,
                const unsigned int* __restrict__ hard,
                float* __restrict__ out)
{
  __shared__ float sm[24];
  const int lane = threadIdx.x & 31;
  const int wv   = threadIdx.x >> 5;
  float hs = 0.f, ps = 0.f, ns = 0.f;
  for (int r = threadIdx.x; r < BS; r += 256) {
    const float p = posd[r];
    const float h = __uint_as_float(hard[r]);
    hs += fmaxf(1.0f + p - h, 0.0f);
    ps += p;
    ns += h;
  }
  hs = waveAllSum(hs); ps = waveAllSum(ps); ns = waveAllSum(ns);
  if (lane == 0) { sm[wv] = hs; sm[8 + wv] = ps; sm[16 + wv] = ns; }
  __syncthreads();
  if (threadIdx.x == 0) {
    float H = 0.f, P = 0.f, Nn = 0.f;
#pragma unroll
    for (int w = 0; w < 8; ++w) { H += sm[w]; P += sm[8 + w]; Nn += sm[16 + w]; }
    out[0] = H / (float)BS;
    out[1] = P / (float)BS;
    out[2] = Nn / (float)BS;
  }
}

extern "C" void kernel_launch(void* const* d_in, const int* in_sizes, int n_in,
                              void* d_out, int out_size, void* d_ws, size_t ws_size,
                              hipStream_t stream) {
  (void)in_sizes; (void)n_in; (void)out_size; (void)ws_size;
  const float* F = (const float*)d_in[0];

  // Workspace layout (needs ~33.7 MB):
  char* ws = (char*)d_ws;
  bf16* A  = (bf16*)ws;                         // 16 MB normalized anchor (bf16)
  bf16* Nb = A + (size_t)BS * DIM;              // 16 MB normalized negative (bf16)
  float* a_sq = (float*)(Nb + (size_t)BS * DIM);
  float* n_sq = a_sq + BS;
  float* posd = n_sq + BS;
  unsigned int* hard = (unsigned int*)(posd + BS);
  float* out = (float*)d_out;

  normalize_kernel<<<BS, 256, 0, stream>>>(F, A, Nb, a_sq, n_sq, posd, hard);
  gemm_min_kernel<<<dim3(BS / 64, BS / 512), 256, 0, stream>>>(A, Nb, a_sq, n_sq, hard);
  finalize_kernel<<<1, 256, 0, stream>>>(posd, hard, out);
}
